// MistralMLP_33260226740847
// MI455X (gfx1250) — matrix operand order repaired and checked by reading, not compiled, not run on hardware
//
#include <hip/hip_runtime.h>

// ---------------------------------------------------------------------------
// MistralMLP with CATS-style neuron selection, tuned for MI455X (gfx1250).
// GEMMs use v_wmma_f32_16x16x32_bf16 (f32 accumulate), f32->bf16 convert on
// the fly while staging tiles in LDS.  All three GEMMs are NT (K-contiguous
// on both operands) which matches the CDNA5 16-bit A/B WMMA VGPR layouts.
// Main loops are register-staged double-buffered: next K-tile's global loads
// are issued before the WMMA burst and consumed (convert->LDS) afterwards,
// so HBM/L2 latency overlaps the matrix pipe.
// ---------------------------------------------------------------------------

typedef __bf16 bf16_t;
typedef bf16_t v8bf  __attribute__((ext_vector_type(8)));
typedef bf16_t v16bf __attribute__((ext_vector_type(16)));
typedef float  v4f   __attribute__((ext_vector_type(4)));
typedef float  v8f   __attribute__((ext_vector_type(8)));

#define LDT 72   // LDS row stride (bf16 elems): 144B -> 16B-aligned frags, bank-staggered

__device__ __forceinline__ bf16_t f2bf(float f) {
  union { float f; unsigned int u; } c; c.f = f;
  unsigned int u = c.u;
  unsigned int r = (u + 0x7FFFu + ((u >> 16) & 1u)) >> 16;   // round-to-nearest-even
  union { unsigned short s; bf16_t b; } o; o.s = (unsigned short)r;
  return o.b;
}

__device__ __forceinline__ void st4bf(bf16_t* d, v4f v) {
  d[0] = f2bf(v[0]); d[1] = f2bf(v[1]); d[2] = f2bf(v[2]); d[3] = f2bf(v[3]);
}

__device__ __forceinline__ v16bf cat8(v8bf lo, v8bf hi) {
  return __builtin_shufflevector(lo, hi, 0,1,2,3,4,5,6,7,8,9,10,11,12,13,14,15);
}

// A-fragment (16x32 bf16): lane m=l%16; lane-half holds K k0..k0+7 & k0+16..k0+23
__device__ __forceinline__ v16bf load_a_frag(const bf16_t* base) {
  return cat8(*(const v8bf*)base, *(const v8bf*)(base + 16));
}
// B-fragment (32x16 bf16): lane n=l%16; lane-half holds K k0..k0+7 & k0+16..k0+23
__device__ __forceinline__ v16bf load_b_frag(const bf16_t* base) {
  return cat8(*(const v8bf*)base, *(const v8bf*)(base + 16));
}

// ---------------------------------------------------------------------------
// Fused gate/up GEMM + SiLU*u epilogue.
// C block 128(M) x 64(I), BK=64, 256 threads = 8 waves, wave tile 32x32,
// dual accumulators (gate & up) share the A (x) fragments.
// Inter[s, i] = silu(x.Wg^T) * (x.Wu^T)
// ---------------------------------------------------------------------------
__global__ __launch_bounds__(256) void gateup_kernel(
    const float* __restrict__ X,  const float* __restrict__ Wg,
    const float* __restrict__ Wu, float* __restrict__ Inter,
    int M, int I, int K)
{
  constexpr int BM = 128, BN = 64, BK = 64;
  __shared__ bf16_t As[BM * LDT];
  __shared__ bf16_t Gs[BN * LDT];
  __shared__ bf16_t Us[BN * LDT];

  const int tid  = threadIdx.x;
  const int m0   = blockIdx.y * BM;
  const int n0   = blockIdx.x * BN;
  const int lane = tid & 31, wid = tid >> 5;
  const int wm   = (wid & 3) * 32;   // wave M offset (4 waves along M)
  const int wn   = (wid >> 2) * 32;  // wave N offset (2 waves along N)
  const int lm   = lane & 15, lh = lane >> 4;
  const int c4   = tid & 15;         // float4 column in 64-wide K tile
  const int rw   = tid >> 4;         // 16 rows per load pass

  v8f zero = {};
  v8f ag[2][2], au[2][2];
  #pragma unroll
  for (int i = 0; i < 2; i++)
    #pragma unroll
    for (int j = 0; j < 2; j++) { ag[i][j] = zero; au[i][j] = zero; }

  // Register staging for double buffering
  v4f ra[8], rg[4], ru[4];
  #pragma unroll
  for (int p = 0; p < 8; p++)
    ra[p] = *(const v4f*)(X + (size_t)(m0 + rw + p * 16) * K + c4 * 4);
  #pragma unroll
  for (int p = 0; p < 4; p++) {
    rg[p] = *(const v4f*)(Wg + (size_t)(n0 + rw + p * 16) * K + c4 * 4);
    ru[p] = *(const v4f*)(Wu + (size_t)(n0 + rw + p * 16) * K + c4 * 4);
  }

  for (int k0 = 0; k0 < K; k0 += BK) {
    // convert + commit staged tile to LDS (waits on the in-flight loads here)
    #pragma unroll
    for (int p = 0; p < 8; p++)
      st4bf(&As[(rw + p * 16) * LDT + c4 * 4], ra[p]);
    #pragma unroll
    for (int p = 0; p < 4; p++) {
      st4bf(&Gs[(rw + p * 16) * LDT + c4 * 4], rg[p]);
      st4bf(&Us[(rw + p * 16) * LDT + c4 * 4], ru[p]);
    }
    __syncthreads();

    // issue next tile's global loads; consumed only after the WMMA burst
    if (k0 + BK < K) {
      const int kn = k0 + BK;
      #pragma unroll
      for (int p = 0; p < 8; p++)
        ra[p] = *(const v4f*)(X + (size_t)(m0 + rw + p * 16) * K + kn + c4 * 4);
      #pragma unroll
      for (int p = 0; p < 4; p++) {
        rg[p] = *(const v4f*)(Wg + (size_t)(n0 + rw + p * 16) * K + kn + c4 * 4);
        ru[p] = *(const v4f*)(Wu + (size_t)(n0 + rw + p * 16) * K + kn + c4 * 4);
      }
      if (k0 + 2 * BK < K) {  // pull tile k0+2*BK toward L2 (global_prefetch_b8)
        __builtin_prefetch(X  + (size_t)(m0 + rw) * K + k0 + 2 * BK + c4 * 4, 0, 1);
        __builtin_prefetch(Wg + (size_t)(n0 + (rw & 63)) * K + k0 + 2 * BK + c4 * 4, 0, 1);
        __builtin_prefetch(Wu + (size_t)(n0 + (rw & 63)) * K + k0 + 2 * BK + c4 * 4, 0, 1);
      }
    }

    #pragma unroll
    for (int ks = 0; ks < 2; ks++) {       // two 16x16x32 WMMA K-steps per tile
      const int ka = ks * 32 + lh * 8;
      const int kb = ks * 32 + lh * 8;
      v16bf a[2], bg[2], bu[2];
      #pragma unroll
      for (int i = 0; i < 2; i++)
        a[i]  = load_a_frag(&As[(wm + i * 16 + lm) * LDT + ka]);
      #pragma unroll
      for (int j = 0; j < 2; j++) {
        bg[j] = load_b_frag(&Gs[(wn + j * 16 + lm) * LDT + kb]);
        bu[j] = load_b_frag(&Us[(wn + j * 16 + lm) * LDT + kb]);
      }
      #pragma unroll
      for (int i = 0; i < 2; i++)
        #pragma unroll
        for (int j = 0; j < 2; j++) {
          ag[i][j] = __builtin_amdgcn_wmma_f32_16x16x32_bf16(
              false, a[i], false, bg[j], (short)0, ag[i][j], false, false);
          au[i][j] = __builtin_amdgcn_wmma_f32_16x16x32_bf16(
              false, a[i], false, bu[j], (short)0, au[i][j], false, false);
        }
    }
    __syncthreads();
  }

  // Epilogue: int_states = silu(g) * u   (C layout: lane l, vgpr v -> M=v+8*lh, N=lm)
  #pragma unroll
  for (int i = 0; i < 2; i++)
    #pragma unroll
    for (int j = 0; j < 2; j++) {
      int row0 = m0 + wm + i * 16 + lh * 8;
      int col  = n0 + wn + j * 16 + lm;
      #pragma unroll
      for (int v = 0; v < 8; v++) {
        float g = ag[i][j][v], u = au[i][j][v];
        float s = g / (1.0f + __expf(-g));
        Inter[(size_t)(row0 + v) * I + col] = s * u;
      }
    }
}

// ---------------------------------------------------------------------------
// Generic NT WMMA GEMM: C[m,n] = sum_k A[m,k]*B[n,k]  (f32 in, bf16 compute).
// Used for the down projection (A = int_states, B = Wd). 128x128 tile, BK=64.
// ---------------------------------------------------------------------------
__global__ __launch_bounds__(256) void gemm_nt_kernel(
    const float* __restrict__ A, const float* __restrict__ B,
    float* __restrict__ C, int M, int N, int K)
{
  constexpr int BM = 128, BN = 128, BK = 64;
  __shared__ bf16_t As[BM * LDT];
  __shared__ bf16_t Bs[BN * LDT];

  const int tid  = threadIdx.x;
  const int m0   = blockIdx.y * BM;
  const int n0   = blockIdx.x * BN;
  const int lane = tid & 31, wid = tid >> 5;
  const int wm   = (wid & 3) * 32;   // 4 waves along M
  const int wn   = (wid >> 2) * 64;  // 2 waves along N, 64 wide each
  const int lm   = lane & 15, lh = lane >> 4;
  const int c4   = tid & 15;
  const int rw   = tid >> 4;

  v8f zero = {};
  v8f acc[2][4];
  #pragma unroll
  for (int i = 0; i < 2; i++)
    #pragma unroll
    for (int j = 0; j < 4; j++) acc[i][j] = zero;

  // Register staging for double buffering
  v4f ra[8], rb[8];
  #pragma unroll
  for (int p = 0; p < 8; p++) {
    ra[p] = *(const v4f*)(A + (size_t)(m0 + rw + p * 16) * K + c4 * 4);
    rb[p] = *(const v4f*)(B + (size_t)(n0 + rw + p * 16) * K + c4 * 4);
  }

  for (int k0 = 0; k0 < K; k0 += BK) {
    // convert + commit staged tile to LDS
    #pragma unroll
    for (int p = 0; p < 8; p++) {
      st4bf(&As[(rw + p * 16) * LDT + c4 * 4], ra[p]);
      st4bf(&Bs[(rw + p * 16) * LDT + c4 * 4], rb[p]);
    }
    __syncthreads();

    // issue next tile's global loads; consumed only after the WMMA burst
    if (k0 + BK < K) {
      const int kn = k0 + BK;
      #pragma unroll
      for (int p = 0; p < 8; p++) {
        ra[p] = *(const v4f*)(A + (size_t)(m0 + rw + p * 16) * K + kn + c4 * 4);
        rb[p] = *(const v4f*)(B + (size_t)(n0 + rw + p * 16) * K + kn + c4 * 4);
      }
      if (k0 + 2 * BK < K) {
        __builtin_prefetch(A + (size_t)(m0 + rw) * K + k0 + 2 * BK + c4 * 4, 0, 1);
        __builtin_prefetch(B + (size_t)(n0 + rw) * K + k0 + 2 * BK + c4 * 4, 0, 1);
      }
    }

    #pragma unroll
    for (int ks = 0; ks < 2; ks++) {
      const int ka = ks * 32 + lh * 8;
      const int kb = ks * 32 + lh * 8;
      v16bf a[2], b[4];
      #pragma unroll
      for (int i = 0; i < 2; i++)
        a[i] = load_a_frag(&As[(wm + i * 16 + lm) * LDT + ka]);
      #pragma unroll
      for (int j = 0; j < 4; j++)
        b[j] = load_b_frag(&Bs[(wn + j * 16 + lm) * LDT + kb]);
      #pragma unroll
      for (int i = 0; i < 2; i++)
        #pragma unroll
        for (int j = 0; j < 4; j++)
          acc[i][j] = __builtin_amdgcn_wmma_f32_16x16x32_bf16(
              false, a[i], false, b[j], (short)0, acc[i][j], false, false);
    }
    __syncthreads();
  }

  #pragma unroll
  for (int i = 0; i < 2; i++)
    #pragma unroll
    for (int j = 0; j < 4; j++) {
      int row0 = m0 + wm + i * 16 + lh * 8;
      int col  = n0 + wn + j * 16 + lm;
      #pragma unroll
      for (int v = 0; v < 8; v++)
        C[(size_t)(row0 + v) * N + col] = acc[i][j][v];
    }
}

// ---------------------------------------------------------------------------
// Per-token inverse squared norm: tokinv[s] = 1 / sum_i Inter[s,i]^2
// ---------------------------------------------------------------------------
__global__ __launch_bounds__(256) void rownorm_kernel(
    const float* __restrict__ Inter, float* __restrict__ tokinv, int I)
{
  __shared__ float red[256];
  const int s = blockIdx.x;
  float ss = 0.0f;
  for (int i = threadIdx.x; i < I; i += 256) {
    float v = Inter[(size_t)s * I + i];
    ss += v * v;
  }
  red[threadIdx.x] = ss;
  __syncthreads();
  for (int o = 128; o > 0; o >>= 1) {
    if (threadIdx.x < o) red[threadIdx.x] += red[threadIdx.x + o];
    __syncthreads();
  }
  if (threadIdx.x == 0) tokinv[s] = 1.0f / red[0];
}

// ---------------------------------------------------------------------------
// Per-neuron statistic (squared; sqrt is monotone so top-k is unchanged):
// nstat[i] = sum_s Inter[s,i]^2 * tokinv[s].  Thread-per-column, coalesced.
// ---------------------------------------------------------------------------
__global__ __launch_bounds__(256) void colstat_kernel(
    const float* __restrict__ Inter, const float* __restrict__ tokinv,
    float* __restrict__ nstat, int M, int I)
{
  __shared__ float tinv[2048];
  for (int s = threadIdx.x; s < M; s += 256) tinv[s] = tokinv[s];
  __syncthreads();
  const int i = blockIdx.x * 256 + threadIdx.x;
  float acc = 0.0f;
  for (int s = 0; s < M; s++) {
    float v = Inter[(size_t)s * I + i];
    acc += v * v * tinv[s];
  }
  nstat[i] = acc;
}

// ---------------------------------------------------------------------------
// Top-k via single-workgroup bitonic sort of 16384 padded (val, idx) pairs in
// 128 KB dynamic LDS (CDNA5: 320 KB per WGP).  Descending; ties -> lower idx.
// ---------------------------------------------------------------------------
__global__ __launch_bounds__(1024) void topk_kernel(
    const float* __restrict__ nstat, int* __restrict__ topk_idx, int n, int k)
{
  constexpr int NP = 16384;
  extern __shared__ unsigned char smem[];
  float* val = (float*)smem;
  int*   idx = (int*)(smem + NP * sizeof(float));

  for (int i = threadIdx.x; i < NP; i += 1024) {
    val[i] = (i < n) ? nstat[i] : -3.0e38f;
    idx[i] = i;
  }
  __syncthreads();

  for (int kk = 2; kk <= NP; kk <<= 1) {
    for (int j = kk >> 1; j > 0; j >>= 1) {
      for (int i = threadIdx.x; i < NP; i += 1024) {
        int ixj = i ^ j;
        if (ixj > i) {
          float va = val[i], vb = val[ixj];
          int   ia = idx[i], ib = idx[ixj];
          bool a_first = (va > vb) || (va == vb && ia < ib);
          bool desc = ((i & kk) == 0);
          if (desc ? !a_first : a_first) {
            val[i] = vb; val[ixj] = va;
            idx[i] = ib; idx[ixj] = ia;
          }
        }
      }
      __syncthreads();
    }
  }
  for (int i = threadIdx.x; i < k; i += 1024) topk_idx[i] = idx[i];
}

// ---------------------------------------------------------------------------
// Row gather: out{g,u}[j,:] = W{g,u}[idx[j],:]   (float4 copies)
// ---------------------------------------------------------------------------
__global__ __launch_bounds__(256) void gather_rows_kernel(
    const float* __restrict__ Wg, const float* __restrict__ Wu,
    const int* __restrict__ idx, float* __restrict__ outg,
    float* __restrict__ outu, int H)
{
  const int j = blockIdx.x;
  const int src = idx[j];
  const v4f* sg = (const v4f*)(Wg + (size_t)src * H);
  const v4f* su = (const v4f*)(Wu + (size_t)src * H);
  v4f* dg = (v4f*)(outg + (size_t)j * H);
  v4f* du = (v4f*)(outu + (size_t)j * H);
  const int n4 = H >> 2;
  for (int t = threadIdx.x; t < n4; t += 256) { dg[t] = sg[t]; du[t] = su[t]; }
}

// ---------------------------------------------------------------------------
// Column gather: outd[h, j] = Wd[h, idx[j]]
// ---------------------------------------------------------------------------
__global__ __launch_bounds__(256) void gather_cols_kernel(
    const float* __restrict__ Wd, const int* __restrict__ idx,
    float* __restrict__ outd, int I, int k)
{
  const int h = blockIdx.x;
  const float* src = Wd + (size_t)h * I;
  float* dst = outd + (size_t)h * k;
  for (int j = threadIdx.x; j < k; j += 256) dst[j] = src[idx[j]];
}

// ---------------------------------------------------------------------------
extern "C" void kernel_launch(void* const* d_in, const int* in_sizes, int n_in,
                              void* d_out, int out_size, void* d_ws, size_t ws_size,
                              hipStream_t stream) {
  const float* x  = (const float*)d_in[0];   // (1, 2048, 4096)
  const float* Wg = (const float*)d_in[1];   // (14336, 4096)
  const float* Wu = (const float*)d_in[2];   // (14336, 4096)
  const float* Wd = (const float*)d_in[3];   // (4096, 14336)

  const int M = 2048, H = 4096, I = 14336;
  const int Kred = I / 2;                    // K_FACTOR = 0.5 -> 7168

  // Workspace layout
  float* inter  = (float*)d_ws;                      // 2048 x 14336 f32
  float* tokinv = inter + (size_t)M * I;             // 2048
  float* nstat  = tokinv + M;                        // 14336
  int*   tidx   = (int*)(nstat + I);                 // 7168

  // Output layout (concatenated flat, f32)
  float* out      = (float*)d_out;
  float* out_down = out;                             // 2048 x 4096
  float* out_wg   = out_down + (size_t)M * H;        // 7168 x 4096
  float* out_wu   = out_wg + (size_t)Kred * H;       // 7168 x 4096
  float* out_wd   = out_wu + (size_t)Kred * H;       // 4096 x 7168

  dim3 blk(256);

  // 1) fused gate/up GEMM + silu*u  -> inter
  gateup_kernel<<<dim3(I / 64, M / 128), blk, 0, stream>>>(x, Wg, Wu, inter, M, I, H);

  // 2) token norms, 3) neuron statistic
  rownorm_kernel<<<dim3(M), blk, 0, stream>>>(inter, tokinv, I);
  colstat_kernel<<<dim3(I / 256), blk, 0, stream>>>(inter, tokinv, nstat, M, I);

  // 4) top-k indices (bitonic sort, 128 KB LDS)
  topk_kernel<<<dim3(1), dim3(1024), 131072, stream>>>(nstat, tidx, I, Kred);

  // 5) weight gathers
  gather_rows_kernel<<<dim3(Kred), blk, 0, stream>>>(Wg, Wu, tidx, out_wg, out_wu, H);
  gather_cols_kernel<<<dim3(H), blk, 0, stream>>>(Wd, tidx, out_wd, I, Kred);

  // 6) down projection GEMM: down[s,h] = sum_i inter[s,i]*Wd[h,i]
  gemm_nt_kernel<<<dim3(H / 128, M / 128), blk, 0, stream>>>(inter, Wd, out_down, M, H, I);
}
